// TPSRoIExtractor_64793876627949
// MI455X (gfx1250) — matrix operand order repaired and checked by reading, not compiled, not run on hardware
//
#include <hip/hip_runtime.h>
#include <math.h>

// ---------------------------------------------------------------------------
// Problem constants (from reference)
// ---------------------------------------------------------------------------
#define BATCH   2
#define CIN     256
#define COUT    256
#define HH      256
#define WW      256
#define NROI    64
#define KROI    (BATCH * NROI)      // 128
#define FPTS    14
#define NPTS    256                 // OUT_H*OUT_W = 8*32

// d_ws layout (bytes)
#define Y_OFF      ((size_t)0)                         // f32 conv output: 134217728 B
#define WBH_OFF    ((size_t)134217728)                 // bf16 weights hi: 9*256*256*2 = 1179648 B
#define WBL_OFF    (WBH_OFF + (size_t)1179648)         // bf16 weights lo
#define IDC_OFF    (WBL_OFF + (size_t)1179648)         // inv_delta_C 17*17 f32 (pad 4096)
#define PHAT_OFF   (IDC_OFF + (size_t)4096)            // P_hat 256*17 f32 (pad 32768)
#define CRD_OFF    (PHAT_OFF + (size_t)32768)          // 6 arrays of 32768 * 4 B
#define CRD_N      (KROI * NPTS)                       // 32768

typedef __attribute__((ext_vector_type(16))) __bf16 v16bf;
typedef __attribute__((ext_vector_type(8)))  float  v8f;

// ---------------------------------------------------------------------------
// 1) Weight prep: conv_w (Cout,Cin,3,3) f32 ->
//    W2[tap][chunk(8)][hf(2)][cout(256)][i(16)] bf16 hi/lo, cin = chunk*32+(i>>3)*16+hf*8+(i&7)
//    => one lane's B-fragment = 16 contiguous bf16 (32B), coalesced across lanes.
// ---------------------------------------------------------------------------
__global__ __launch_bounds__(256) void prep_weights(const float* __restrict__ w,
                                                    __bf16* __restrict__ wbh,
                                                    __bf16* __restrict__ wbl) {
  int e = blockIdx.x * 256 + threadIdx.x;   // grid 2304 blocks covers 9*65536 exactly
  int i     = e & 15;
  int cout  = (e >> 4) & 255;
  int hf    = (e >> 12) & 1;
  int chunk = (e >> 13) & 7;
  int tap   = e >> 16;
  int cin   = (chunk << 5) + ((i >> 3) << 4) + (hf << 3) + (i & 7);
  float v = w[((cout << 8) + cin) * 9 + tap];
  __bf16 hi = (__bf16)v;
  wbh[e] = hi;
  wbl[e] = (__bf16)(v - (float)hi);
}

// ---------------------------------------------------------------------------
// 2) TPS constants: inv_delta_C (17x17) via f64 Gauss-Jordan, P_hat (256x17)
// ---------------------------------------------------------------------------
__global__ __launch_bounds__(256) void tps_constants(float* __restrict__ invDC,
                                                     float* __restrict__ phat) {
  __shared__ double M[17][34];
  __shared__ double Cx[FPTS], Cy[FPTS];
  int tid = threadIdx.x;
  if (tid == 0) {
    for (int i = 0; i < 7; ++i) {
      double cx = -1.0 + 2.0 * (double)i / 6.0;
      Cx[i] = cx;     Cy[i] = -1.0;
      Cx[i + 7] = cx; Cy[i + 7] = 1.0;
    }
    for (int i = 0; i < 17; ++i)
      for (int j = 0; j < 34; ++j)
        M[i][j] = (j == 17 + i) ? 1.0 : 0.0;
    for (int i = 0; i < FPTS; ++i) {
      M[i][0] = 1.0; M[i][1] = Cx[i]; M[i][2] = Cy[i];
      for (int j = 0; j < FPTS; ++j) {
        double dx = Cx[i] - Cx[j], dy = Cy[i] - Cy[j];
        double d = sqrt(dx * dx + dy * dy);
        if (i == j) d = 1.0;
        M[i][3 + j] = d * d * log(d);
      }
    }
    for (int j = 0; j < FPTS; ++j) {
      M[14][3 + j] = Cx[j];
      M[15][3 + j] = Cy[j];
      M[16][3 + j] = 1.0;
    }
    for (int col = 0; col < 17; ++col) {
      int piv = col; double best = fabs(M[col][col]);
      for (int r = col + 1; r < 17; ++r) {
        double v = fabs(M[r][col]);
        if (v > best) { best = v; piv = r; }
      }
      if (piv != col)
        for (int j = 0; j < 34; ++j) { double t = M[col][j]; M[col][j] = M[piv][j]; M[piv][j] = t; }
      double inv = 1.0 / M[col][col];
      for (int j = 0; j < 34; ++j) M[col][j] *= inv;
      for (int r = 0; r < 17; ++r) {
        if (r == col) continue;
        double f = M[r][col];
        if (f != 0.0)
          for (int j = 0; j < 34; ++j) M[r][j] -= f * M[col][j];
      }
    }
    for (int i = 0; i < 17; ++i)
      for (int j = 0; j < 17; ++j)
        invDC[i * 17 + j] = (float)M[i][17 + j];
  }
  __syncthreads();
  {
    double Px = (2.0 * (double)(tid & 31) - 31.0) / 32.0;
    double Py = (2.0 * (double)(tid >> 5) - 7.0) / 8.0;
    phat[tid * 17 + 0] = 1.0f;
    phat[tid * 17 + 1] = (float)Px;
    phat[tid * 17 + 2] = (float)Py;
    for (int j = 0; j < FPTS; ++j) {
      double dx = Px - Cx[j], dy = Py - Cy[j];
      double rn = sqrt(dx * dx + dy * dy);
      phat[tid * 17 + 3 + j] = (float)(rn * rn * log(rn + 1e-6));
    }
  }
}

// ---------------------------------------------------------------------------
// 3) Conv3x3 + bias + BN + ReLU, implicit GEMM, bf16x3-split WMMA.
//    Block 256 thr (8 waves). Tile: M=32 pixels (one h row), N=256 couts.
//    Wave: 2 M-tiles x 2 N-tiles (couts [wave*32, wave*32+32)).
//    Software-pipelined: next chunk's feats loads issued before the tap loop,
//    consumed (convert+LDS store) at the top of the next iteration.
//    LDS tile stored pre-swizzled in A-fragment element order:
//      tile[(r*34+c)*2+hf][i]  (uint pairs of bf16) => fragment = 32B ds read.
//      cin->(hf,i): hf=bit3(cin); k'=cin&~8;
//      g = ((k'>>1)&3)|(((k'>>4)&1)<<2); i = 2g|(k'&1)
// ---------------------------------------------------------------------------
__global__ __launch_bounds__(256) void conv_wmma(
    const float* __restrict__ feats, const __bf16* __restrict__ wbh,
    const __bf16* __restrict__ wbl,  const float* __restrict__ bias,
    const float* __restrict__ gamma, const float* __restrict__ beta,
    const float* __restrict__ mean,  const float* __restrict__ var,
    float* __restrict__ y) {
  __shared__ alignas(32) unsigned tileH[1632];   // 3*34*2 blocks * 8 uints
  __shared__ alignas(32) unsigned tileL[1632];
  int t    = blockIdx.x;                // 4096 = B * H * (W/32)
  int b    = t >> 11;
  int rem  = t & 2047;
  int h    = rem >> 3;
  int w0   = (rem & 7) << 5;
  int tid  = threadIdx.x;
  int wave = tid >> 5;
  int lane = tid & 31;
  int mrow = lane & 15;
  int hf   = lane >> 4;
  int cbase = wave << 5;                // wave * 32 couts

  // ---- precompute staging slots (chunk-invariant): 1632 cin-pairs x (r,c) ----
  int goff[7], loff[7], flag[7];        // flag: 0 none, 1 zero-fill, 2 load
#pragma unroll
  for (int j = 0; j < 7; ++j) {
    int pe = tid + (j << 8);
    flag[j] = 0; goff[j] = 0; loff[j] = 0;
    if (pe < 1632) {
      int cinp = pe / 102;              // 0..15 (cin pair)
      int r2   = pe - cinp * 102;
      int r    = r2 / 34;
      int c    = r2 - r * 34;
      int cin  = cinp << 1;             // even
      int hfk  = (cin >> 3) & 1;
      int kp   = cin & ~8;
      int g    = ((kp >> 1) & 3) | (((kp >> 4) & 1) << 2);
      int i0   = (g << 1) | (kp & 1);   // even
      loff[j]  = ((r * 34 + c) * 2 + hfk) * 8 + (i0 >> 1);
      int hs   = h - 1 + r;
      int wsrc = w0 - 1 + c;
      if ((unsigned)hs < 256u && (unsigned)wsrc < 256u) {
        flag[j] = 2;
        goff[j] = (((b << 8) + cin) * 256 + hs) * 256 + wsrc;
      } else {
        flag[j] = 1;
      }
    }
  }

  // ---- prologue: load chunk 0 feats into registers ----
  float va[7], vb[7];
#pragma unroll
  for (int j = 0; j < 7; ++j) {
    va[j] = 0.f; vb[j] = 0.f;
    if (flag[j] == 2) {
      const float* p = feats + goff[j];
      va[j] = p[0];
      vb[j] = p[65536];
    }
  }

  v8f acc00 = {}, acc01 = {}, acc10 = {}, acc11 = {};

  for (int chunk = 0; chunk < 8; ++chunk) {
    // ---- consume in-flight values: split hi/lo bf16, store swizzled packed ----
#pragma unroll
    for (int j = 0; j < 7; ++j) {
      if (flag[j]) {
        float v0 = va[j], v1 = vb[j];
        __bf16 h0 = (__bf16)v0, h1 = (__bf16)v1;
        __bf16 l0 = (__bf16)(v0 - (float)h0), l1 = (__bf16)(v1 - (float)h1);
        unsigned ph = (unsigned)__builtin_bit_cast(unsigned short, h0) |
                      ((unsigned)__builtin_bit_cast(unsigned short, h1) << 16);
        unsigned pl = (unsigned)__builtin_bit_cast(unsigned short, l0) |
                      ((unsigned)__builtin_bit_cast(unsigned short, l1) << 16);
        tileH[loff[j]] = ph;
        tileL[loff[j]] = pl;
      }
    }
    __syncthreads();

    // ---- issue next chunk's global loads; they complete behind the WMMAs ----
    if (chunk < 7) {
#pragma unroll
      for (int j = 0; j < 7; ++j) {
        if (flag[j] == 2) {
          const float* p = feats + goff[j] + ((chunk + 1) << 21);
          va[j] = p[0];
          vb[j] = p[65536];
        }
      }
    }

    for (int tap = 0; tap < 9; ++tap) {
      int kh = tap / 3;
      int kw = tap - kh * 3;
      int c0 = mrow + kw;               // M-tile 0
      int c1 = c0 + 16;                 // M-tile 1
      int o0 = ((kh * 34 + c0) * 2 + hf) * 8;
      int o1 = ((kh * 34 + c1) * 2 + hf) * 8;
      v16bf ah0 = *(const v16bf*)&tileH[o0];
      v16bf al0 = *(const v16bf*)&tileL[o0];
      v16bf ah1 = *(const v16bf*)&tileH[o1];
      v16bf al1 = *(const v16bf*)&tileL[o1];
      const v16bf* wph = (const v16bf*)wbh + (((((tap << 3) + chunk) << 1) + hf) << 8);
      const v16bf* wpl = (const v16bf*)wbl + (((((tap << 3) + chunk) << 1) + hf) << 8);
      {
        v16bf bh = wph[cbase + mrow];
        v16bf bl = wpl[cbase + mrow];
        acc00 = __builtin_amdgcn_wmma_f32_16x16x32_bf16(false, ah0, false, bh, (short)0, acc00, false, false);
        acc00 = __builtin_amdgcn_wmma_f32_16x16x32_bf16(false, ah0, false, bl, (short)0, acc00, false, false);
        acc00 = __builtin_amdgcn_wmma_f32_16x16x32_bf16(false, al0, false, bh, (short)0, acc00, false, false);
        acc10 = __builtin_amdgcn_wmma_f32_16x16x32_bf16(false, ah1, false, bh, (short)0, acc10, false, false);
        acc10 = __builtin_amdgcn_wmma_f32_16x16x32_bf16(false, ah1, false, bl, (short)0, acc10, false, false);
        acc10 = __builtin_amdgcn_wmma_f32_16x16x32_bf16(false, al1, false, bh, (short)0, acc10, false, false);
      }
      {
        v16bf bh = wph[cbase + 16 + mrow];
        v16bf bl = wpl[cbase + 16 + mrow];
        acc01 = __builtin_amdgcn_wmma_f32_16x16x32_bf16(false, ah0, false, bh, (short)0, acc01, false, false);
        acc01 = __builtin_amdgcn_wmma_f32_16x16x32_bf16(false, ah0, false, bl, (short)0, acc01, false, false);
        acc01 = __builtin_amdgcn_wmma_f32_16x16x32_bf16(false, al0, false, bh, (short)0, acc01, false, false);
        acc11 = __builtin_amdgcn_wmma_f32_16x16x32_bf16(false, ah1, false, bh, (short)0, acc11, false, false);
        acc11 = __builtin_amdgcn_wmma_f32_16x16x32_bf16(false, ah1, false, bl, (short)0, acc11, false, false);
        acc11 = __builtin_amdgcn_wmma_f32_16x16x32_bf16(false, al1, false, bh, (short)0, acc11, false, false);
      }
    }
    __syncthreads();
  }

  // ---- epilogue: bias + BN + ReLU; C/D layout M = r + hf*8, N = mrow ----
  auto emit = [&](const v8f& acc, int ti, int mh) {
    int cout = cbase + (ti << 4) + mrow;
    float s  = gamma[cout] * rsqrtf(var[cout] + 1e-5f);
    float bb = bias[cout];
    float mm = mean[cout];
    float be = beta[cout];
#pragma unroll
    for (int r = 0; r < 8; ++r) {
      int w = w0 + (mh << 4) + r + (hf << 3);
      float v = (acc[r] + bb - mm) * s + be;
      y[((((b << 8) + cout) << 8) + h) * 256 + w] = fmaxf(v, 0.f);
    }
  };
  emit(acc00, 0, 0);
  emit(acc01, 1, 0);
  emit(acc10, 0, 1);
  emit(acc11, 1, 1);
}

// ---------------------------------------------------------------------------
// 4) TPS per-ROI sample coordinates (one block per ROI, thread per point)
// ---------------------------------------------------------------------------
__global__ __launch_bounds__(256) void tps_coords(
    const float* __restrict__ fp, const float* __restrict__ invDC,
    const float* __restrict__ phat,
    int* __restrict__ ix0, int* __restrict__ ix1,
    int* __restrict__ iy0, int* __restrict__ iy1,
    float* __restrict__ wxp, float* __restrict__ wyp) {
  int k   = blockIdx.x;
  int tid = threadIdx.x;
  __shared__ float Cp[17][2];
  __shared__ float T[17][2];
  __shared__ float prm[4];   // x1, y1, Wc, Hc
  if (tid == 0) {
    float px[FPTS], py[FPTS];
    float mnx = 1e30f, mxx = -1e30f, mny = 1e30f, mxy = -1e30f;
    for (int f = 0; f < FPTS; ++f) {
      float x  = fp[(k * FPTS + f) * 2 + 0] * 0.25f;
      float yv = fp[(k * FPTS + f) * 2 + 1] * 0.25f;
      x  = fminf(fmaxf(x, 0.f), (float)WW);
      yv = fminf(fmaxf(yv, 0.f), (float)HH);
      px[f] = x; py[f] = yv;
      mnx = fminf(mnx, x); mxx = fmaxf(mxx, x);
      mny = fminf(mny, yv); mxy = fmaxf(mxy, yv);
    }
    float x1 = floorf(mnx), x2 = floorf(mxx) + 1.f;
    float y1 = floorf(mny), y2 = floorf(mxy) + 1.f;
    float x2e = fminf(x2, (float)WW), y2e = fminf(y2, (float)HH);
    for (int f = 0; f < FPTS; ++f) {
      Cp[f][0] = 2.f * (px[f] - x1) / (x2 - x1) - 1.f;
      Cp[f][1] = 2.f * (py[f] - y1) / (y2 - y1) - 1.f;
    }
    for (int f = FPTS; f < 17; ++f) { Cp[f][0] = 0.f; Cp[f][1] = 0.f; }
    prm[0] = x1; prm[1] = y1; prm[2] = x2e - x1; prm[3] = y2e - y1;
  }
  __syncthreads();
  if (tid < 34) {
    int i = tid >> 1, d = tid & 1;
    float s = 0.f;
    for (int j = 0; j < 17; ++j) s += invDC[i * 17 + j] * Cp[j][d];
    T[i][d] = s;
  }
  __syncthreads();
  float ppx = 0.f, ppy = 0.f;
  for (int j = 0; j < 17; ++j) {
    float ph = phat[tid * 17 + j];
    ppx += ph * T[j][0];
    ppy += ph * T[j][1];
  }
  float x1 = prm[0], y1 = prm[1], Wc = prm[2], Hc = prm[3];
  float sx = ((ppx + 1.f) * Wc - 1.f) * 0.5f;
  float sy = ((ppy + 1.f) * Hc - 1.f) * 0.5f;
  sx = fminf(fmaxf(sx, 0.f), Wc - 1.f) + x1;
  sy = fminf(fmaxf(sy, 0.f), Hc - 1.f) + y1;
  float fx0 = floorf(sx), fy0 = floorf(sy);
  float xmax = x1 + Wc - 1.f, ymax = y1 + Hc - 1.f;
  int idx = (k << 8) + tid;
  ix0[idx] = (int)fminf(fmaxf(fx0, x1), xmax);
  ix1[idx] = (int)fminf(fmaxf(fx0 + 1.f, x1), xmax);
  iy0[idx] = (int)fminf(fmaxf(fy0, y1), ymax);
  iy1[idx] = (int)fminf(fmaxf(fy0 + 1.f, y1), ymax);
  wxp[idx] = sx - fx0;
  wyp[idx] = sy - fy0;
}

// ---------------------------------------------------------------------------
// 5) Bilinear gather: block = (k, c), thread = point p
// ---------------------------------------------------------------------------
__global__ __launch_bounds__(256) void sample_kernel(
    const float* __restrict__ y,
    const int* __restrict__ ix0, const int* __restrict__ ix1,
    const int* __restrict__ iy0, const int* __restrict__ iy1,
    const float* __restrict__ wxp, const float* __restrict__ wyp,
    float* __restrict__ out) {
  int blk = blockIdx.x;          // 32768 = K * C
  int k = blk >> 8;
  int c = blk & 255;
  int p = threadIdx.x;
  int idx = (k << 8) + p;
  int b = k >> 6;                // k / NROI
  const float* yc = y + ((size_t)((b << 8) + c) << 16);
  int x0 = ix0[idx], x1 = ix1[idx], y0 = iy0[idx], y1 = iy1[idx];
  float wx = wxp[idx], wy = wyp[idx];
  float v00 = yc[(y0 << 8) + x0];
  float v01 = yc[(y0 << 8) + x1];
  float v10 = yc[(y1 << 8) + x0];
  float v11 = yc[(y1 << 8) + x1];
  float r = v00 * (1.f - wx) * (1.f - wy) + v01 * wx * (1.f - wy)
          + v10 * (1.f - wx) * wy + v11 * wx * wy;
  out[((size_t)blk << 8) + p] = r;
}

// ---------------------------------------------------------------------------
extern "C" void kernel_launch(void* const* d_in, const int* in_sizes, int n_in,
                              void* d_out, int out_size, void* d_ws, size_t ws_size,
                              hipStream_t stream) {
  const float* feats  = (const float*)d_in[0];
  const float* fid    = (const float*)d_in[1];
  const float* conv_w = (const float*)d_in[2];
  const float* conv_b = (const float*)d_in[3];
  const float* bn_g   = (const float*)d_in[4];
  const float* bn_b   = (const float*)d_in[5];
  const float* bn_m   = (const float*)d_in[6];
  const float* bn_v   = (const float*)d_in[7];

  char* ws = (char*)d_ws;
  float*  y     = (float*)(ws + Y_OFF);
  __bf16* wbh   = (__bf16*)(ws + WBH_OFF);
  __bf16* wbl   = (__bf16*)(ws + WBL_OFF);
  float*  invDC = (float*)(ws + IDC_OFF);
  float*  phat  = (float*)(ws + PHAT_OFF);
  int*    ix0   = (int*)(ws + CRD_OFF);
  int*    ix1   = (int*)(ws + CRD_OFF + (size_t)CRD_N * 4);
  int*    iy0   = (int*)(ws + CRD_OFF + (size_t)CRD_N * 8);
  int*    iy1   = (int*)(ws + CRD_OFF + (size_t)CRD_N * 12);
  float*  wx    = (float*)(ws + CRD_OFF + (size_t)CRD_N * 16);
  float*  wy    = (float*)(ws + CRD_OFF + (size_t)CRD_N * 20);
  float*  out   = (float*)d_out;

  prep_weights<<<2304, 256, 0, stream>>>(conv_w, wbh, wbl);
  tps_constants<<<1, 256, 0, stream>>>(invDC, phat);
  conv_wmma<<<BATCH * HH * (WW / 32), 256, 0, stream>>>(feats, wbh, wbl, conv_b,
                                                        bn_g, bn_b, bn_m, bn_v, y);
  tps_coords<<<KROI, 256, 0, stream>>>(fid, invDC, phat, ix0, ix1, iy0, iy1, wx, wy);
  sample_kernel<<<KROI * COUT, 256, 0, stream>>>(y, ix0, ix1, iy0, iy1, wx, wy, out);
}
